// MaxCutPool_73383811219610
// MI455X (gfx1250) — matrix operand order repaired and checked by reading, not compiled, not run on hardware
//
#include <hip/hip_runtime.h>
#include <math.h>

// ---------------------------------------------------------------------------
// MaxCutPool pipeline for MI455X (gfx1250, wave32, WMMA + TDM).
//   N=100000 nodes, E=1600000 edges, C=128 feats, K_TOP=50000.
//   GEMMs -> v_wmma_f32_16x16x32_bf16, fragment-native LDS layout
//     (2x ds_load_b128 per operand, A fragments register-resident).
//   GCN-layer weight tiles -> tensor_load_to_lds (TDM DMA, TENSORcnt),
//     overlapped with A-tile bf16 conversion, joined by s_wait_tensorcnt.
//   Graph ops -> templated-width atomics; working set (~100MB) fits in L2.
// ---------------------------------------------------------------------------

#define N_NODES 100000
#define N_EDGES 1600000
#define CFEAT   128
#define K_TOP   50000
#define DELTA_W 2.0f
#define BETA_W  1.0f
#define MAX_IT  10

typedef __attribute__((ext_vector_type(16))) __bf16 v16bf;
typedef __attribute__((ext_vector_type(8)))  float  v8f;

#if __has_builtin(__builtin_amdgcn_tensor_load_to_lds) && __has_builtin(__builtin_amdgcn_s_wait_tensorcnt)
#define CDNA5_HAS_TDM 1
#else
#define CDNA5_HAS_TDM 0
#endif

#if CDNA5_HAS_TDM
typedef __attribute__((ext_vector_type(4))) unsigned int u32x4;
typedef __attribute__((ext_vector_type(8))) int          i32x8;
typedef __attribute__((ext_vector_type(4))) int          i32x4;

// Issue a TDM load of a 2D f32 tensor tile (whole dim0 x dim1 matrix) into LDS.
// D# layout per CDNA5 ISA ch.8: group0 = {count, lds_addr, global_addr, type=2},
// group1 = {data_size=4B, tensor_dim0/1, tile_dim0/1, tensor_dim0_stride}.
// Groups 2/3 zero (2D tensor). Tracked by TENSORcnt.
__device__ __forceinline__ void tdm_load_2d_f32(const float* gsrc, void* lds_dst,
                                                unsigned dim0, unsigned dim1,
                                                unsigned stride0) {
  const unsigned long long ga = (unsigned long long)(size_t)gsrc;
  const unsigned lds = (unsigned)(size_t)lds_dst;  // LDS aperture: addr[31:0]
  u32x4 g0;
  g0[0] = 1u;                                        // count=1, user mode
  g0[1] = lds;                                       // lds_addr [63:32]
  g0[2] = (unsigned)(ga & 0xFFFFFFFFu);              // global_addr lo [95:64]
  g0[3] = (unsigned)((ga >> 32) & 0x01FFFFFFu) | (2u << 30);  // addr hi | type=2
  i32x8 g1;
  g1[0] = (int)(2u << 16);                           // data_size=2 (4 bytes)
  g1[1] = (int)(dim0 << 16);                         // tensor_dim0[15:0] @ bit48
  g1[2] = (int)((dim0 >> 16) | (dim1 << 16));        // dim0 hi | tensor_dim1 lo
  g1[3] = (int)((dim1 >> 16) | (dim0 << 16));        // dim1 hi | tile_dim0
  g1[4] = (int)(dim1 & 0xFFFFu);                     // tile_dim1 (tile_dim2=0)
  g1[5] = (int)stride0;                              // tensor_dim0_stride lo32
  g1[6] = 0;                                         // stride hi | dim1_stride lo
  g1[7] = 0;
  i32x4 g2 = {0, 0, 0, 0}, g3 = {0, 0, 0, 0};
#if __clang_major__ >= 23
  i32x8 g4 = {0, 0, 0, 0, 0, 0, 0, 0};
  __builtin_amdgcn_tensor_load_to_lds(g0, g1, g2, g3, g4, 0);
#else
  __builtin_amdgcn_tensor_load_to_lds(g0, g1, g2, g3, 0);
#endif
}
#endif  // CDNA5_HAS_TDM

__device__ __forceinline__ unsigned short f32_to_bf16_bits(float f) {
  unsigned u = __builtin_bit_cast(unsigned, f);
  u += 0x7FFFu + ((u >> 16) & 1u);            // round to nearest even
  return (unsigned short)(u >> 16);
}

// ---------------------------------------------------------------------------
// WMMA tile GEMM:  Cout[nrows x NOUT] = A[nrows x K] @ W[K x NOUT] (+bias)
// Block = 256 threads = 8 waves; block computes a 128-row tile, each wave a
// 16-row strip. LDS holds bf16 operands in *fragment order*:
//   As[rowTile][kstep][lane][e0..15], Bs[colTile][kstep][lane][e0..15]
// For small-K layers the raw f32 W tile is DMA'd into LDS by the TDM while
// the A tile is being converted; the B conversion then reads LDS, not global.
// ---------------------------------------------------------------------------
template <int K, int NOUT, bool HAS_BIAS>
__global__ __launch_bounds__(256) void wmma_gemm_kernel(
    const float* __restrict__ A, const float* __restrict__ W,
    const float* __restrict__ bias, float* __restrict__ Cout, int nrows) {
  constexpr int KP     = (K + 31) & ~31;
  constexpr int NP     = (NOUT + 15) & ~15;
  constexpr int KSTEPS = KP / 32;
  constexpr int NTILES = NP / 16;
  constexpr bool KFULL = (K == KP);
  constexpr bool NFULL = ((NOUT & 15) == 0);
  constexpr bool WANT_TDM = (KSTEPS == 1);   // small weight tile (<= 4KB)

  __shared__ unsigned short As[8 * KSTEPS * 32 * 16];      // fragment order
  __shared__ unsigned short Bs[NTILES * KSTEPS * 32 * 16]; // fragment order
#if CDNA5_HAS_TDM
  __shared__ float Wraw[WANT_TDM ? (K * NOUT) : 1];
#endif

  const int tid  = threadIdx.x;
  const int row0 = blockIdx.x * 128;

#if CDNA5_HAS_TDM
  // Kick off TDM DMA of the raw weight tile; overlaps with A conversion below.
  if (WANT_TDM && tid < 32)
    tdm_load_2d_f32(W, &Wraw[0], (unsigned)NOUT, (unsigned)K, (unsigned)NOUT);
#endif

  // Prefetch next row tile (emits global_prefetch_b8).
  if (tid == 0 && row0 + 128 < nrows)
    __builtin_prefetch(&A[(size_t)(row0 + 128) * K], 0, 1);

  // ---- stage A fragments: one slot = one lane's 16 bf16 elements ----------
  // ISA 16-bit A layout: lane(0-15) row M=lane, VGPR v<4 -> K=khalf+2v(+1),
  // v>=4 -> K=16+khalf+2(v-4)(+1); khalf = 0 (lanes 0-15) / 8 (lanes 16-31).
  for (int s = tid; s < 8 * KSTEPS * 32; s += 256) {
    const int lane = s & 31;
    const int ks   = (s >> 5) % KSTEPS;
    const int rt   = s / (32 * KSTEPS);
    const int m     = lane & 15;
    const int khalf = (lane & 16) ? 8 : 0;
    const int gr    = row0 + rt * 16 + m;
    union { unsigned short u16[16]; uint4 q[2]; } tmp;
    if (gr < nrows) {
#pragma unroll
      for (int v = 0; v < 8; ++v) {
        const int ka = (v < 4) ? (khalf + 2 * v) : (16 + khalf + 2 * (v - 4));
        const int c  = ks * 32 + ka;
        float f0 = 0.0f, f1 = 0.0f;
        if (KFULL || c < K)     f0 = A[(size_t)gr * K + c];
        if (KFULL || c + 1 < K) f1 = A[(size_t)gr * K + c + 1];
        tmp.u16[2 * v]     = f32_to_bf16_bits(f0);
        tmp.u16[2 * v + 1] = f32_to_bf16_bits(f1);
      }
    } else {
      tmp.q[0] = make_uint4(0, 0, 0, 0);
      tmp.q[1] = make_uint4(0, 0, 0, 0);
    }
    uint4* dq = (uint4*)&As[s * 16];
    dq[0] = tmp.q[0];
    dq[1] = tmp.q[1];
  }

#if CDNA5_HAS_TDM
  if (WANT_TDM) {
    if (tid < 32) __builtin_amdgcn_s_wait_tensorcnt(0);  // join TDM DMA
    __syncthreads();                                     // publish Wraw
  }
#endif

  // ---- stage B fragments --------------------------------------------------
  // ISA 16-bit B layout: lane holds column n=lane&15; K = khalf + e (e<8),
  // 16 + khalf + (e-8) (e>=8); khalf = 0 (lanes 0-15) / 8 (lanes 16-31).
  for (int s = tid; s < NTILES * KSTEPS * 32; s += 256) {
    const int lane = s & 31;
    const int ks   = (s >> 5) % KSTEPS;
    const int ct   = s / (32 * KSTEPS);
    const int n     = lane & 15;
    const int khalf = (lane & 16) ? 8 : 0;
    const int gc    = ct * 16 + n;
    union { unsigned short u16[16]; uint4 q[2]; } tmp;
#pragma unroll
    for (int e = 0; e < 16; ++e) {
      const int k = ks * 32 + khalf + e + ((e >= 8) ? 8 : 0);
      float f = 0.0f;
      if ((KFULL || k < K) && (NFULL || gc < NOUT)) {
#if CDNA5_HAS_TDM
        f = WANT_TDM ? Wraw[k * NOUT + gc] : W[k * NOUT + gc];
#else
        f = W[k * NOUT + gc];
#endif
      }
      tmp.u16[e] = f32_to_bf16_bits(f);
    }
    uint4* dq = (uint4*)&Bs[s * 16];
    dq[0] = tmp.q[0];
    dq[1] = tmp.q[1];
  }
  __syncthreads();

  const int wave  = tid >> 5;
  const int lane  = tid & 31;
  const int mlane = lane & 15;

  // A fragments for this wave's 16-row strip: load once, keep in VGPRs.
  v16bf afr[KSTEPS];
#pragma unroll
  for (int ks = 0; ks < KSTEPS; ++ks)
    afr[ks] = *(const v16bf*)&As[((wave * KSTEPS + ks) * 32 + lane) * 16];

  const bool fulltile = (row0 + 128 <= nrows);

#pragma unroll
  for (int nt = 0; nt < NTILES; ++nt) {
    v8f acc = {};
#pragma unroll
    for (int ks = 0; ks < KSTEPS; ++ks) {
      v16bf b = *(const v16bf*)&Bs[((nt * KSTEPS + ks) * 32 + lane) * 16];
      acc = __builtin_amdgcn_wmma_f32_16x16x32_bf16(
          false, afr[ks], false, b, (short)0, acc, false, false);
    }
    // D layout: VGPR r -> M = r (lanes 0-15) / 8+r (lanes 16-31), N = lane&15
    const int gc = nt * 16 + mlane;
    const bool colok = NFULL || (gc < NOUT);
    float bv = 0.0f;
    if (HAS_BIAS && colok) bv = bias[gc];
    const int grb = row0 + wave * 16 + ((lane < 16) ? 0 : 8);
    if (colok) {
      if (fulltile) {
#pragma unroll
        for (int r = 0; r < 8; ++r)
          Cout[(size_t)(grb + r) * NOUT + gc] = acc[r] + bv;
      } else {
#pragma unroll
        for (int r = 0; r < 8; ++r)
          if (grb + r < nrows) Cout[(size_t)(grb + r) * NOUT + gc] = acc[r] + bv;
      }
    }
  }
}

// ------------------------- utility fills -----------------------------------
__global__ void fill_f32(float* p, float v, long long n) {
  long long i = (long long)blockIdx.x * blockDim.x + threadIdx.x;
  if (i < n) p[i] = v;
}
__global__ void fill_i32(int* p, int v, long long n) {
  long long i = (long long)blockIdx.x * blockDim.x + threadIdx.x;
  if (i < n) p[i] = v;
}

// ------------------------- graph normalization -----------------------------
__global__ void deg_kernel(const int* __restrict__ src,
                           const float* __restrict__ ew,
                           float* __restrict__ deg, int e) {
  int i = blockIdx.x * blockDim.x + threadIdx.x;
  if (i < e) atomicAdd(&deg[src[i]], ew[i]);
}
__global__ void dinv_kernel(const float* __restrict__ deg,
                            float* __restrict__ dinv, int n) {
  int i = blockIdx.x * blockDim.x + threadIdx.x;
  if (i < n) {
    float d = deg[i];
    dinv[i] = (d > 0.0f) ? rsqrtf(fmaxf(d, 1e-12f)) : 0.0f;
  }
}
__global__ void wnorm_kernel(const int* __restrict__ src,
                             const int* __restrict__ dst,
                             const float* __restrict__ ew,
                             const float* __restrict__ dinv,
                             float* __restrict__ wn, int e) {
  int i = blockIdx.x * blockDim.x + threadIdx.x;
  if (i < e) wn[i] = ew[i] * dinv[src[i]] * dinv[dst[i]];
}

// ------------------------- GCN aggregation ---------------------------------
// One thread per edge: indices/weight loaded once, source row read as float4,
// unrolled global_atomic_add_f32 stream into the destination row.
template <int UNITS>
__global__ void scatter_kernel(const int* __restrict__ src,
                               const int* __restrict__ dst,
                               const float* __restrict__ wn,
                               const float* __restrict__ hw,
                               float* __restrict__ agg, int e) {
  int ed = blockIdx.x * blockDim.x + threadIdx.x;
  if (ed >= e) return;
  const int s = src[ed], d = dst[ed];
  const float w = DELTA_W * wn[ed];
  const float4* hs = (const float4*)&hw[(size_t)s * UNITS];
  float* ad = &agg[(size_t)d * UNITS];
#pragma unroll
  for (int q = 0; q < UNITS / 4; ++q) {
    float4 v = hs[q];
    atomicAdd(&ad[4 * q + 0], w * v.x);
    atomicAdd(&ad[4 * q + 1], w * v.y);
    atomicAdd(&ad[4 * q + 2], w * v.z);
    atomicAdd(&ad[4 * q + 3], w * v.w);
  }
}
template <int UNITS>
__global__ void gcn_finish_kernel(const float* __restrict__ agg,
                                  const float* __restrict__ hw,
                                  const float* __restrict__ b,
                                  float* __restrict__ hout, int n) {
  int i = blockIdx.x * blockDim.x + threadIdx.x;
  if (i >= n) return;
  const float4* ag = (const float4*)&agg[(size_t)i * UNITS];
  const float4* hv = (const float4*)&hw[(size_t)i * UNITS];
  const float4* bb = (const float4*)b;
  float4* ho = (float4*)&hout[(size_t)i * UNITS];
#pragma unroll
  for (int q = 0; q < UNITS / 4; ++q) {
    float4 a = ag[q], h = hv[q], bq = bb[q], r;
    r.x = tanhf(a.x + (1.0f - DELTA_W) * h.x + bq.x);
    r.y = tanhf(a.y + (1.0f - DELTA_W) * h.y + bq.y);
    r.z = tanhf(a.z + (1.0f - DELTA_W) * h.z + bq.z);
    r.w = tanhf(a.w + (1.0f - DELTA_W) * h.w + bq.w);
    ho[q] = r;
  }
}

// ------------------------- MLP head + score --------------------------------
__global__ void mlp_final_kernel(const float* __restrict__ h,  // N x 8
                                 const float* __restrict__ w1, const float* __restrict__ b1,
                                 const float* __restrict__ w2, const float* __restrict__ b2,
                                 const float* __restrict__ fw, const float* __restrict__ fb,
                                 float* __restrict__ z, float* __restrict__ z_out, int n) {
  int i = blockIdx.x * blockDim.x + threadIdx.x;
  if (i >= n) return;
  float hin[8];
#pragma unroll
  for (int c = 0; c < 8; ++c) hin[c] = h[(size_t)i * 8 + c];
  float t1[16];
#pragma unroll
  for (int o = 0; o < 16; ++o) {
    float s = b1[o];
#pragma unroll
    for (int c = 0; c < 8; ++c) s += hin[c] * w1[c * 16 + o];
    t1[o] = fmaxf(s, 0.0f);
  }
  float t2[16];
#pragma unroll
  for (int o = 0; o < 16; ++o) {
    float s = b2[o];
#pragma unroll
    for (int c = 0; c < 16; ++c) s += t1[c] * w2[c * 16 + o];
    t2[o] = fmaxf(s, 0.0f);
  }
  float s = fb[0];
#pragma unroll
  for (int c = 0; c < 16; ++c) s += t2[c] * fw[c];
  float zz = tanhf(s);
  z[i] = zz;
  z_out[i] = zz;
}

// ------------------------- aux loss ----------------------------------------
__global__ void aux_loss_kernel(const float* __restrict__ ew,
                                const int* __restrict__ src,
                                const int* __restrict__ dst,
                                const float* __restrict__ z,
                                float* __restrict__ out, int e) {
  __shared__ float red[256];
  int i = blockIdx.x * 256 + threadIdx.x;
  float v = 0.0f;
  if (i < e) v = ew[i] * z[src[i]] * z[dst[i]];
  red[threadIdx.x] = v;
  __syncthreads();
  for (int s = 128; s > 0; s >>= 1) {
    if (threadIdx.x < s) red[threadIdx.x] += red[threadIdx.x + s];
    __syncthreads();
  }
  if (threadIdx.x == 0) atomicAdd(out, red[0] * (BETA_W / (float)e));
}

// ------------------------- top-k via histogram -----------------------------
__device__ __forceinline__ int z_bin(float z) {
  int b = (int)((z + 1.0f) * 128.0f);
  return b < 0 ? 0 : (b > 255 ? 255 : b);
}
__global__ void hist_kernel(const float* __restrict__ z, int* __restrict__ hist, int n) {
  int i = blockIdx.x * blockDim.x + threadIdx.x;
  if (i < n) atomicAdd(&hist[z_bin(z[i])], 1);
}
__global__ void thresh_kernel(const int* __restrict__ hist, int* __restrict__ tinfo, int k) {
  if (blockIdx.x == 0 && threadIdx.x == 0) {
    int cum = 0, t = 0;
    for (int b = 255; b >= 0; --b) {
      cum += hist[b];
      if (cum >= k) { t = b; break; }
    }
    tinfo[0] = t;
  }
}
__global__ void select_kernel(const float* __restrict__ z,
                              const int* __restrict__ tinfo, int* __restrict__ ctr,
                              int* __restrict__ kept, int* __restrict__ clus,
                              int n, int k) {
  int i = blockIdx.x * blockDim.x + threadIdx.x;
  if (i >= n) return;
  if (z_bin(z[i]) >= tinfo[0]) {
    int id = atomicAdd(ctr, 1);
    if (id < k) { kept[id] = i; clus[i] = id; }
  }
}

// ------------------------- label propagation -------------------------------
__global__ void prop_kernel(const int* __restrict__ src, const int* __restrict__ dst,
                            const int* __restrict__ cur, int* __restrict__ prop, int e) {
  int i = blockIdx.x * blockDim.x + threadIdx.x;
  if (i >= e) return;
  atomicMax(&prop[dst[i]], cur[src[i]]);
  atomicMax(&prop[src[i]], cur[dst[i]]);
}
__global__ void merge_kernel(int* __restrict__ cur, const int* __restrict__ prop, int n) {
  int i = blockIdx.x * blockDim.x + threadIdx.x;
  if (i < n) { int c = cur[i]; cur[i] = (c >= 0) ? c : prop[i]; }
}
__global__ void clamp_cluster_kernel(int* __restrict__ cur, int* __restrict__ out, int n) {
  int i = blockIdx.x * blockDim.x + threadIdx.x;
  if (i < n) { int c = cur[i]; c = c < 0 ? 0 : c; cur[i] = c; out[i] = c; }
}

// ------------------------- outputs -----------------------------------------
__global__ void edge_out_kernel(const int* __restrict__ src, const int* __restrict__ dst,
                                const int* __restrict__ clus, const float* __restrict__ ew,
                                int* __restrict__ nei, float* __restrict__ newW, int e) {
  int i = blockIdx.x * blockDim.x + threadIdx.x;
  if (i >= e) return;
  int cs = clus[src[i]], cd = clus[dst[i]];
  nei[i] = cs;
  nei[e + i] = cd;
  newW[i] = (cs == cd) ? 0.0f : ew[i];
}
__global__ void xpool_kernel(const float* __restrict__ xe, const float* __restrict__ z,
                             const int* __restrict__ kept, float* __restrict__ xp,
                             int k) {
  long long idx = (long long)blockIdx.x * blockDim.x + threadIdx.x;
  if (idx >= (long long)k * CFEAT) return;
  int kk = (int)(idx >> 7), cc = (int)(idx & 127);   // CFEAT == 128
  int i = kept[kk];
  xp[idx] = xe[(size_t)i * CFEAT + cc] * z[i];
}

// ---------------------------------------------------------------------------
extern "C" void kernel_launch(void* const* d_in, const int* in_sizes, int n_in,
                              void* d_out, int out_size, void* d_ws, size_t ws_size,
                              hipStream_t stream) {
  (void)in_sizes; (void)n_in; (void)out_size; (void)ws_size;

  const int N = N_NODES, E = N_EDGES, C = CFEAT, KTOP = K_TOP;

  // --- inputs (setup_inputs dict order, lists flattened in order) ---
  const float* x      = (const float*)d_in[0];
  const int*   ei     = (const int*)  d_in[1];
  const float* ew     = (const float*)d_in[2];
  const float* emb_w  = (const float*)d_in[3];
  const float* emb_b  = (const float*)d_in[4];
  const float* gw[12]; const float* gb[12];
  for (int i = 0; i < 12; ++i) { gw[i] = (const float*)d_in[5 + i]; gb[i] = (const float*)d_in[17 + i]; }
  const float* mw1 = (const float*)d_in[29];
  const float* mw2 = (const float*)d_in[30];
  const float* mb1 = (const float*)d_in[31];
  const float* mb2 = (const float*)d_in[32];
  const float* fw  = (const float*)d_in[33];
  const float* fb  = (const float*)d_in[34];
  const int* src = ei;
  const int* dst = ei + E;

  // --- outputs, concatenated flat in return order ---
  float* out_f   = (float*)d_out;
  float* o_xpool = out_f;                                    // K*C
  int*   o_nei   = (int*)(out_f + (size_t)KTOP * C);         // 2E
  float* o_neww  = out_f + (size_t)KTOP * C + 2 * (size_t)E; // E
  int*   o_kept  = (int*)(o_neww + E);                       // K
  float* o_z     = (float*)(o_kept + KTOP);                  // N
  float* o_aux   = o_z + N;                                  // 1
  int*   o_clus  = (int*)(o_aux + 1);                        // N

  // --- workspace carve (~98 MB) ---
  char* wsp = (char*)d_ws;
  auto carve = [&](size_t bytes) -> void* {
    void* p = (void*)wsp;
    wsp += (bytes + 255) & ~(size_t)255;
    return p;
  };
  float* xe    = (float*)carve((size_t)N * C * 4);
  float* h0    = (float*)carve((size_t)N * 32 * 4);
  float* h1    = (float*)carve((size_t)N * 32 * 4);   // hw buffer
  float* agg   = (float*)carve((size_t)N * 32 * 4);
  float* deg   = (float*)carve((size_t)N * 4);
  float* dinv  = (float*)carve((size_t)N * 4);
  float* wn    = (float*)carve((size_t)E * 4);
  float* zbuf  = (float*)carve((size_t)N * 4);
  int*   clus  = (int*)  carve((size_t)N * 4);
  int*   prop  = (int*)  carve((size_t)N * 4);
  int*   hist  = (int*)  carve(256 * 4);
  int*   tinfo = (int*)  carve(64 * 4);
  int*   ctr   = (int*)  carve(64 * 4);

  auto blocks = [](long long n) { return (unsigned)((n + 255) / 256); };
  const unsigned gemm_grid = (unsigned)((N + 127) / 128);

  // 1) symmetric degree normalization
  fill_f32<<<blocks(N), 256, 0, stream>>>(deg, 0.0f, N);
  deg_kernel<<<blocks(E), 256, 0, stream>>>(src, ew, deg, E);
  dinv_kernel<<<blocks(N), 256, 0, stream>>>(deg, dinv, N);
  wnorm_kernel<<<blocks(E), 256, 0, stream>>>(src, dst, ew, dinv, wn, E);

  // 2) embedding GEMM (WMMA): xe = x @ emb_w + emb_b
  wmma_gemm_kernel<128, 128, true><<<gemm_grid, 256, 0, stream>>>(x, emb_w, emb_b, xe, N);

  // 3) 12 GCN layers: hw = h@W (WMMA, W via TDM); agg = scatter(...); tanh
#define GCN_LAYER(L, KIN, KOUT, AIN)                                            \
  wmma_gemm_kernel<KIN, KOUT, false><<<gemm_grid, 256, 0, stream>>>(AIN, gw[L], nullptr, h1, N); \
  fill_f32<<<blocks((long long)N * KOUT), 256, 0, stream>>>(agg, 0.0f, (long long)N * KOUT); \
  scatter_kernel<KOUT><<<blocks(E), 256, 0, stream>>>(src, dst, wn, h1, agg, E); \
  gcn_finish_kernel<KOUT><<<blocks(N), 256, 0, stream>>>(agg, h1, gb[L], h0, N);

  GCN_LAYER(0, 128, 32, xe)
  GCN_LAYER(1, 32, 32, h0)
  GCN_LAYER(2, 32, 32, h0)
  GCN_LAYER(3, 32, 32, h0)
  GCN_LAYER(4, 32, 16, h0)
  GCN_LAYER(5, 16, 16, h0)
  GCN_LAYER(6, 16, 16, h0)
  GCN_LAYER(7, 16, 16, h0)
  GCN_LAYER(8, 16, 8, h0)
  GCN_LAYER(9, 8, 8, h0)
  GCN_LAYER(10, 8, 8, h0)
  GCN_LAYER(11, 8, 8, h0)
#undef GCN_LAYER

  // 4) MLP head -> score z (written to ws and to output)
  mlp_final_kernel<<<blocks(N), 256, 0, stream>>>(h0, mw1, mb1, mw2, mb2, fw, fb,
                                                  zbuf, o_z, N);

  // 5) aux loss = BETA * mean(ew * z[src] * z[dst])
  fill_f32<<<1, 256, 0, stream>>>(o_aux, 0.0f, 1);
  aux_loss_kernel<<<blocks(E), 256, 0, stream>>>(ew, src, dst, zbuf, o_aux, E);

  // 6) top-K selection via 256-bin histogram of z in (-1,1)
  fill_i32<<<1, 256, 0, stream>>>(hist, 0, 256);
  fill_i32<<<1, 256, 0, stream>>>(ctr, 0, 1);
  hist_kernel<<<blocks(N), 256, 0, stream>>>(zbuf, hist, N);
  thresh_kernel<<<1, 64, 0, stream>>>(hist, tinfo, KTOP);
  fill_i32<<<blocks(N), 256, 0, stream>>>(clus, -1, N);
  fill_i32<<<blocks(KTOP), 256, 0, stream>>>(o_kept, 0, KTOP);
  select_kernel<<<blocks(N), 256, 0, stream>>>(zbuf, tinfo, ctr, o_kept, clus, N, KTOP);

  // 7) cluster label propagation (MAX_ITER rounds of edge max)
  for (int it = 0; it < MAX_IT; ++it) {
    fill_i32<<<blocks(N), 256, 0, stream>>>(prop, -1, N);
    prop_kernel<<<blocks(E), 256, 0, stream>>>(src, dst, clus, prop, E);
    merge_kernel<<<blocks(N), 256, 0, stream>>>(clus, prop, N);
  }
  clamp_cluster_kernel<<<blocks(N), 256, 0, stream>>>(clus, o_clus, N);

  // 8) coarsened edges + pooled features
  edge_out_kernel<<<blocks(E), 256, 0, stream>>>(src, dst, clus, ew, o_nei, o_neww, E);
  fill_f32<<<blocks((long long)KTOP * C), 256, 0, stream>>>(o_xpool, 0.0f, (long long)KTOP * C);
  xpool_kernel<<<blocks((long long)KTOP * C), 256, 0, stream>>>(xe, zbuf, o_kept, o_xpool, KTOP);
}
